// LayerNormLSTM_4483945857412
// MI455X (gfx1250) — matrix operand order repaired and checked by reading, not compiled, not run on hardware
//
#include <hip/hip_runtime.h>
#include <hip/hip_bf16.h>
#include <stdint.h>

// ---------------- problem sizes (compile-time) ----------------
#define SEQ    128
#define BATCH  64
#define IN_DIM 1024
#define HID    2048
#define G4     (4*HID)        // 8192 gate width
#define NTILES (G4/16)        // 512 WMMA column tiles
#define KT_IH  (IN_DIM/32)    // 32  k-tiles for input GEMM
#define KT_HH  (HID/32)       // 64  k-tiles for recurrent GEMM
#define KSPLIT 2              // split-K factor for the step GEMMs
#define LN_EPS 1e-5f

typedef __attribute__((ext_vector_type(16))) __bf16 v16bf;
typedef __attribute__((ext_vector_type(8)))  float  v8f;

// ---------------- helpers ----------------
__device__ __forceinline__ uint16_t f2bf(float f) {
  uint32_t u = __float_as_uint(f);
  uint32_t r = u + 0x7FFFu + ((u >> 16) & 1u);   // round-to-nearest-even
  return (uint16_t)(r >> 16);
}
__device__ __forceinline__ uint32_t pack2bf(float lo, float hi) {
  return (uint32_t)f2bf(lo) | ((uint32_t)f2bf(hi) << 16);
}
__device__ __forceinline__ float sigmoidf_(float x) {
  return 1.0f / (1.0f + __expf(-x));
}

// reduce two values across a 256-thread block
__device__ __forceinline__ void block_reduce2(float& a, float& b,
                                              float* r1, float* r2) {
  int t = threadIdx.x;
  r1[t] = a; r2[t] = b;
  __syncthreads();
  for (int off = 128; off > 0; off >>= 1) {
    if (t < off) { r1[t] += r1[t + off]; r2[t] += r2[t + off]; }
    __syncthreads();
  }
  a = r1[0]; b = r2[0];
  __syncthreads();   // safe reuse of r1/r2
}

// ---------------- prologue kernels ----------------
// Pack W [G4, K] (row-major) into WMMA B-fragment layout:
// Wp[nTile][kTile][lane][8] u32, lane<16 -> K 0..15 of tile, lane>=16 -> K 16..31,
// each u32 = (bf16 K=2v, bf16 K=2v+1), N = nTile*16 + (lane&15).  B[k][n] = W[n][k].
__global__ void pack_w_kernel(const float* __restrict__ W,
                              uint32_t* __restrict__ Wp,
                              int K, int kTiles) {
  int idx = blockIdx.x * 256 + threadIdx.x;
  int total = NTILES * kTiles * 32 * 8;
  if (idx >= total) return;
  int v    = idx & 7;
  int lane = (idx >> 3) & 31;
  int kt   = (idx >> 8) % kTiles;
  int nt   = (idx >> 8) / kTiles;
  int n    = nt * 16 + (lane & 15);
  int half = lane >> 4;
  int k    = kt * 32 + half * 16 + 2 * v;
  Wp[idx] = pack2bf(W[(size_t)n * K + k], W[(size_t)n * K + k + 1]);
}

__global__ void conv_bf16_kernel(const float* __restrict__ src,
                                 uint16_t* __restrict__ dst, int total) {
  int i = blockIdx.x * 256 + threadIdx.x;
  if (i < total) dst[i] = f2bf(src[i]);
}

__global__ void zero_kernel(uint32_t* __restrict__ p, int n) {
  int i = blockIdx.x * 256 + threadIdx.x;
  if (i < n) p[i] = 0u;
}

// ---------------- K1: dual gate GEMM (WMMA bf16 -> f32, split-K) ----------------
// grid = (G4/256, 2 gemms, KSPLIT). block = 256 threads = 8 waves.
// Each wave owns 32 columns (two 16-wide N-tiles) and all 4 M-tiles (M=64),
// software-pipelined over its K-range with double-buffered fragments.
__global__ __launch_bounds__(256)
void gemm_gates_kernel(const uint16_t* __restrict__ actIH,   // [64, IN_DIM] bf16
                       const uint32_t* __restrict__ WpIH,
                       const uint16_t* __restrict__ actHH,   // [64, HID] bf16
                       const uint32_t* __restrict__ WpHH,
                       float* __restrict__ preIH,            // [KSPLIT][64][G4]
                       float* __restrict__ preHH) {
  const bool isHH = (blockIdx.y != 0);
  const int  ks   = blockIdx.z;
  const uint16_t* act = isHH ? actHH : actIH;
  const uint32_t* Wp  = isHH ? WpHH  : WpIH;
  float* out = (isHH ? preHH : preIH) + (size_t)ks * BATCH * G4;
  const int K         = isHH ? HID   : IN_DIM;
  const int kTilesTot = isHH ? KT_HH : KT_IH;
  const int ktHalf    = kTilesTot / KSPLIT;       // 32 or 16 (even)
  const int ktBase    = ks * ktHalf;

  const int wave = threadIdx.x >> 5;
  const int lane = threadIdx.x & 31;
  const int n0   = blockIdx.x * 256 + wave * 32;  // two n-tiles per wave
  const int half = lane >> 4;
  const int mrow = lane & 15;

  const uint32_t* wbase0 =
      Wp + ((size_t)(n0 >> 4) * kTilesTot + ktBase) * 256 + (size_t)lane * 8;
  const uint32_t* wbase1 = wbase0 + (size_t)kTilesTot * 256;  // second n-tile

  union F { uint32_t u[8]; v16bf v; uint4 q[2]; };

  auto loadB = [&](F& b0, F& b1, int kt) {
    const uint4* p0 = (const uint4*)(wbase0 + (size_t)kt * 256);
    const uint4* p1 = (const uint4*)(wbase1 + (size_t)kt * 256);
    b0.q[0] = p0[0]; b0.q[1] = p0[1];
    b1.q[0] = p1[0]; b1.q[1] = p1[1];
  };
  auto loadA = [&](F* a, int kt) {
    const int kb = (ktBase + kt) * 32 + half * 16;  // contiguous K halves (match packed B)
    #pragma unroll
    for (int mt = 0; mt < 4; ++mt) {
      const uint16_t* ar = act + (size_t)(mt * 16 + mrow) * K + kb;
      a[mt].q[0] = *(const uint4*)(ar);
      a[mt].q[1] = *(const uint4*)(ar + 8);
    }
  };

  v8f acc[4][2] = {};
  auto mma = [&](F* a, F& b0, F& b1) {
    #pragma unroll
    for (int mt = 0; mt < 4; ++mt) {
      acc[mt][0] = __builtin_amdgcn_wmma_f32_16x16x32_bf16(
          false, a[mt].v, false, b0.v, (short)0, acc[mt][0], false, false);
      acc[mt][1] = __builtin_amdgcn_wmma_f32_16x16x32_bf16(
          false, a[mt].v, false, b1.v, (short)0, acc[mt][1], false, false);
    }
  };

  // double-buffered, software-pipelined main loop (ktHalf is even)
  F a0[4], a1[4], b00, b01, b10, b11;
  loadA(a0, 0);
  loadB(b00, b01, 0);
  for (int kt = 0; kt < ktHalf; kt += 2) {
    loadA(a1, kt + 1);                       // issue next-tile loads first
    loadB(b10, b11, kt + 1);
    mma(a0, b00, b01);                       // consume current while next in flight
    if (kt + 2 < ktHalf) {
      loadA(a0, kt + 2);
      loadB(b00, b01, kt + 2);
    }
    mma(a1, b10, b11);
  }

  // C/D layout: VGPR r -> M=r (lanes<16) / M=8+r (lanes>=16); N striped on lanes
  #pragma unroll
  for (int nt2 = 0; nt2 < 2; ++nt2) {
    const int col = n0 + nt2 * 16 + mrow;
    #pragma unroll
    for (int mt = 0; mt < 4; ++mt) {
      #pragma unroll
      for (int r = 0; r < 8; ++r) {
        int row = mt * 16 + r + half * 8;
        out[(size_t)row * G4 + col] = acc[mt][nt2][r];
      }
    }
  }
}

// ---------------- K2: fused LN + gates + cell + LN(c) + h + y ----------------
// one 256-thread block per batch row; sums the KSPLIT partial pre-activations
__global__ __launch_bounds__(256)
void step_pointwise_kernel(const float* __restrict__ preIH,
                           const float* __restrict__ preHH,
                           const float* __restrict__ b_ih,
                           const float* __restrict__ b_hh,
                           const float* __restrict__ g_ih,
                           const float* __restrict__ be_ih,
                           const float* __restrict__ g_hh,
                           const float* __restrict__ be_hh,
                           const float* __restrict__ g_ho,
                           const float* __restrict__ be_ho,
                           const float* __restrict__ W_out,
                           const float* __restrict__ b_out,
                           float* __restrict__ cbuf,
                           float* __restrict__ hbuf,
                           uint16_t* __restrict__ hb,
                           float* __restrict__ y, int s) {
  const int b = blockIdx.x;
  const int t = threadIdx.x;
  const size_t KS = (size_t)BATCH * G4;          // split-K partial stride
  const float* uIH0 = preIH + (size_t)b * G4;
  const float* uIH1 = uIH0 + KS;
  const float* uHH0 = preHH + (size_t)b * G4;
  const float* uHH1 = uHH0 + KS;

  __shared__ float red1[256];
  __shared__ float red2[256];

  // LN stats over the 8192-wide pre-activations (bias folded in)
  float s1 = 0.f, s2 = 0.f, s3 = 0.f, s4 = 0.f;
  for (int n = t; n < G4; n += 256) {
    float a = uIH0[n] + uIH1[n] + b_ih[n];
    float c = uHH0[n] + uHH1[n] + b_hh[n];
    s1 += a; s2 += a * a; s3 += c; s4 += c * c;
  }
  block_reduce2(s1, s2, red1, red2);
  const float muI = s1 * (1.0f / G4);
  const float rsI = rsqrtf(s2 * (1.0f / G4) - muI * muI + LN_EPS);
  block_reduce2(s3, s4, red1, red2);
  const float muH = s3 * (1.0f / G4);
  const float rsH = rsqrtf(s4 * (1.0f / G4) - muH * muH + LN_EPS);

  auto gate = [&](int n) -> float {
    float a = (uIH0[n] + uIH1[n] + b_ih[n] - muI) * rsI * g_ih[n] + be_ih[n];
    float c = (uHH0[n] + uHH1[n] + b_hh[n] - muH) * rsH * g_hh[n] + be_hh[n];
    return a + c;
  };

  // gates + cell update (thread t owns hidden units j = t + 256*i)
  float cn[8], ov[8];
  float cs1 = 0.f, cs2 = 0.f;
  #pragma unroll
  for (int i = 0; i < 8; ++i) {
    int j = t + i * 256;
    float gi = sigmoidf_(gate(j));
    float gf = sigmoidf_(gate(HID + j));
    float go = sigmoidf_(gate(2 * HID + j));
    float gg = tanhf(gate(3 * HID + j));
    float cv = gf * cbuf[(size_t)b * HID + j] + gi * gg;
    cn[i] = cv; ov[i] = go;
    cs1 += cv; cs2 += cv * cv;
  }
  block_reduce2(cs1, cs2, red1, red2);
  const float muC = cs1 * (1.0f / HID);
  const float rsC = rsqrtf(cs2 * (1.0f / HID) - muC * muC + LN_EPS);

  float yd = 0.f;
  #pragma unroll
  for (int i = 0; i < 8; ++i) {
    int j = t + i * 256;
    float hn = ov[i] * tanhf((cn[i] - muC) * rsC * g_ho[j] + be_ho[j]);
    cbuf[(size_t)b * HID + j] = cn[i];
    hbuf[(size_t)b * HID + j] = hn;
    hb[(size_t)b * HID + j]   = f2bf(hn);   // bf16 copy for next-step GEMM
    yd += hn * W_out[j];
  }
  float dummy = 0.f;
  block_reduce2(yd, dummy, red1, red2);
  if (t == 0) y[b * SEQ + s] = sigmoidf_(yd + b_out[0]);
}

// ---------------- host: launch sequence (graph-capture safe) ----------------
extern "C" void kernel_launch(void* const* d_in, const int* in_sizes, int n_in,
                              void* d_out, int out_size, void* d_ws, size_t ws_size,
                              hipStream_t stream) {
  const float* x     = (const float*)d_in[0];
  const float* W_ih  = (const float*)d_in[1];
  const float* b_ih  = (const float*)d_in[2];
  const float* W_hh  = (const float*)d_in[3];
  const float* b_hh  = (const float*)d_in[4];
  const float* g_ih  = (const float*)d_in[5];
  const float* be_ih = (const float*)d_in[6];
  const float* g_hh  = (const float*)d_in[7];
  const float* be_hh = (const float*)d_in[8];
  const float* g_ho  = (const float*)d_in[9];
  const float* be_ho = (const float*)d_in[10];
  const float* W_out = (const float*)d_in[11];
  const float* b_out = (const float*)d_in[12];
  float* out = (float*)d_out;   // y[64,128] | hy[64,2048] | cy[64,2048]

  uint8_t* ws = (uint8_t*)d_ws;
  size_t off = 0;
  auto alloc = [&](size_t bytes) -> uint8_t* {
    uint8_t* p = ws + off;
    off += (bytes + 255) & ~(size_t)255;
    return p;
  };
  uint32_t* WpIH  = (uint32_t*)alloc((size_t)NTILES * KT_IH * 256 * 4); // 16.8 MB
  uint32_t* WpHH  = (uint32_t*)alloc((size_t)NTILES * KT_HH * 256 * 4); // 33.6 MB
  uint16_t* xb    = (uint16_t*)alloc((size_t)SEQ * BATCH * IN_DIM * 2); // 16.8 MB
  uint16_t* hb    = (uint16_t*)alloc((size_t)BATCH * HID * 2);          // 256 KB
  float*    hbuf  = (float*)   alloc((size_t)BATCH * HID * 4);          // 512 KB
  float*    cbuf  = (float*)   alloc((size_t)BATCH * HID * 4);          // 512 KB
  float*    preIH = (float*)   alloc((size_t)KSPLIT * BATCH * G4 * 4);  // 4 MB
  float*    preHH = (float*)   alloc((size_t)KSPLIT * BATCH * G4 * 4);  // 4 MB
  (void)in_sizes; (void)n_in; (void)out_size; (void)ws_size;

  // prologue: weight packing, x conversion, state zeroing
  {
    int tot = NTILES * KT_IH * 256;
    pack_w_kernel<<<tot / 256, 256, 0, stream>>>(W_ih, WpIH, IN_DIM, KT_IH);
  }
  {
    int tot = NTILES * KT_HH * 256;
    pack_w_kernel<<<tot / 256, 256, 0, stream>>>(W_hh, WpHH, HID, KT_HH);
  }
  {
    int tot = SEQ * BATCH * IN_DIM;
    conv_bf16_kernel<<<tot / 256, 256, 0, stream>>>(x, xb, tot);
  }
  {
    // hb, hbuf, cbuf are contiguous (256B-aligned, sizes multiple of 256B)
    int nu32 = (int)(((size_t)BATCH * HID * 2 + (size_t)BATCH * HID * 8) / 4);
    zero_kernel<<<(nu32 + 255) / 256, 256, 0, stream>>>((uint32_t*)hb, nu32);
  }

  // recurrence
  for (int s = 0; s < SEQ; ++s) {
    const uint16_t* xs = xb + (size_t)s * BATCH * IN_DIM;
    dim3 g1(G4 / 256, 2, KSPLIT);   // 32 x 2 x 2 = 128 blocks per step
    gemm_gates_kernel<<<g1, 256, 0, stream>>>(xs, WpIH, hb, WpHH, preIH, preHH);
    step_pointwise_kernel<<<BATCH, 256, 0, stream>>>(
        preIH, preHH, b_ih, b_hh, g_ih, be_ih, g_hh, be_hh, g_ho, be_ho,
        W_out, b_out, cbuf, hbuf, hb, out, s);
  }

  // hy / cy
  hipMemcpyAsync(out + BATCH * SEQ, hbuf, (size_t)BATCH * HID * 4,
                 hipMemcpyDeviceToDevice, stream);
  hipMemcpyAsync(out + BATCH * SEQ + BATCH * HID, cbuf, (size_t)BATCH * HID * 4,
                 hipMemcpyDeviceToDevice, stream);
}
